// CausalSelfAttention_18760417149593
// MI455X (gfx1250) — matrix operand order repaired and checked by reading, not compiled, not run on hardware
//
#include <hip/hip_runtime.h>
#include <hip/hip_bf16.h>

typedef __bf16 bf16;
typedef __attribute__((ext_vector_type(16))) __bf16 v16bf;
typedef __attribute__((ext_vector_type(8)))  __bf16 v8bf;
typedef __attribute__((ext_vector_type(8)))  float  v8f;
typedef __attribute__((ext_vector_type(4)))  unsigned v4u;
typedef __attribute__((ext_vector_type(8)))  int      v8i;
typedef __attribute__((ext_vector_type(4)))  int      v4i;

#define DEV __device__ __forceinline__

DEV bf16 f2b(float f) {
  unsigned u = __builtin_bit_cast(unsigned, f);
  u += 0x7FFFu + ((u >> 16) & 1u);          // round-to-nearest-even
  unsigned short h = (unsigned short)(u >> 16);
  return __builtin_bit_cast(bf16, h);
}
DEV float b2f(bf16 b) {
  unsigned short h = __builtin_bit_cast(unsigned short, b);
  unsigned u = (unsigned)h << 16;
  return __builtin_bit_cast(float, u);
}

union Frag { v16bf v; v8bf h[2]; };
union Acc  { v8f  v; float f[8]; };

DEV v16bf load_frag(const bf16* p0, const bf16* p1) {
  Frag f;
  f.h[0] = *reinterpret_cast<const v8bf*>(p0);
  f.h[1] = *reinterpret_cast<const v8bf*>(p1);
  return f.v;
}

#define WMMA_BF16(A_, B_, C_) \
  __builtin_amdgcn_wmma_f32_16x16x32_bf16(false, (A_), false, (B_), (short)0, (C_), false, false)

// DPP16 row-rotate butterfly reduction over a 16-lane half (rows stay in-half
// in the WMMA C layout). Entirely VALU: no ds_bpermute / dscnt waits.
#define DPP_ROR(x_, n_)                                                         \
  __builtin_bit_cast(float, __builtin_amdgcn_update_dpp(                        \
      0, __builtin_bit_cast(int, (x_)), 0x120 | (n_), 0xF, 0xF, true))

DEV float half_max(float x) {
  x = fmaxf(x, DPP_ROR(x, 8));
  x = fmaxf(x, DPP_ROR(x, 4));
  x = fmaxf(x, DPP_ROR(x, 2));
  x = fmaxf(x, DPP_ROR(x, 1));
  return x;
}
DEV float half_sum(float x) {
  x += DPP_ROR(x, 8);
  x += DPP_ROR(x, 4);
  x += DPP_ROR(x, 2);
  x += DPP_ROR(x, 1);
  return x;
}

// LDS byte offset of a __shared__ object (generic -> addrspace(3) -> int)
typedef __attribute__((address_space(3))) char* lds_cptr;
DEV unsigned lds_off(const void* p) {
  return (unsigned)(unsigned long long)(lds_cptr)(char*)p;
}

// TDM: 2D tile load Global->LDS. tensor dims == tile dims (no OOB), 2-byte elems.
// D# per CDNA5 ISA 8.3-8.6: group0 {count,lds_addr,global_addr,type=2},
// group1 {data_size=1 @ [17:16], dim0@[79:48], dim1@[111:80], tile0@[127:112],
//         tile1@[143:128], stride0@[207:160]}, groups2/3 = 0 (<=2D).
DEV void tdm_load_2d(unsigned lds_byte, const void* gptr,
                     unsigned dim0, unsigned rows, unsigned stride0) {
  unsigned long long ga = (unsigned long long)gptr;
  v4u g0;
  g0[0] = 1u;                                   // count=1, user mode
  g0[1] = lds_byte;                             // lds_addr
  g0[2] = (unsigned)(ga & 0xFFFFFFFFu);         // global_addr[31:0]
  g0[3] = (unsigned)((ga >> 32) & 0x1FFFFFFu) | (2u << 30);  // addr[56:32], type=2
  v8i g1;
  g1[0] = (int)(1u << 16);                      // workgroup_mask=0, data_size=1 (2B)
  g1[1] = (int)(dim0 << 16);                    // tensor_dim0[15:0]
  g1[2] = (int)((dim0 >> 16) | (rows << 16));   // tensor_dim0[31:16] | tensor_dim1[15:0]
  g1[3] = (int)((rows >> 16) | (dim0 << 16));   // tensor_dim1[31:16] | tile_dim0
  g1[4] = (int)(rows & 0xFFFFu);                // tile_dim1 (tile_dim2 = 0)
  g1[5] = (int)stride0;                         // tensor_dim0_stride[31:0]
  g1[6] = 0;                                    // stride0 hi | stride1 lo
  g1[7] = 0;
  v4i z4 = {0, 0, 0, 0};
  v8i z8 = {0, 0, 0, 0, 0, 0, 0, 0};
  __builtin_amdgcn_tensor_load_to_lds(g0, g1, z4, z4, z8, 0);
}

// ---------------------------------------------------------------- cast f32->bf16
__global__ void cast_f32_bf16(const float* __restrict__ src, bf16* __restrict__ dst, int n) {
  int i = blockIdx.x * blockDim.x + threadIdx.x;
  int stride = gridDim.x * blockDim.x;
  for (; i < n; i += stride) dst[i] = f2b(src[i]);
}

// ---------------------------------------------------------------- GEMM: Out = A @ W^T + bias
// One wave = 32x64 output (2 M-tiles x 4 N-tiles), double-buffered fragments.
template<bool OUT_F32>
__global__ __launch_bounds__(128, 2) void gemm_bf16_wmma(
    const bf16* __restrict__ A, const bf16* __restrict__ W,
    const float* __restrict__ bias, void* __restrict__ Out,
    int M, int N, int K) {
  const int lane = threadIdx.x & 31;
  const int wave = threadIdx.x >> 5;
  const int lh = lane >> 4;
  const int ll = lane & 15;
  const int mlo = (blockIdx.y * 4 + wave) * 32;
  const int nlo = blockIdx.x * 64;

  Acc acc[2][4];
#pragma unroll
  for (int i = 0; i < 2; ++i)
#pragma unroll
    for (int j = 0; j < 4; ++j)
#pragma unroll
      for (int r = 0; r < 8; ++r) acc[i][j].f[r] = 0.f;

  const int aoff0 = (mlo + ll) * K + lh * 8;
  const int aoff1 = aoff0 + 16 * K;
  int woff[4];
#pragma unroll
  for (int j = 0; j < 4; ++j) woff[j] = (nlo + j * 16 + ll) * K + lh * 8;

  v16bf af0 = load_frag(A + aoff0, A + aoff0 + 16);
  v16bf af1 = load_frag(A + aoff1, A + aoff1 + 16);
  v16bf wf[4];
#pragma unroll
  for (int j = 0; j < 4; ++j) wf[j] = load_frag(W + woff[j], W + woff[j] + 16);

  for (int kk = 32; kk < K; kk += 32) {
    v16bf af0n = load_frag(A + aoff0 + kk, A + aoff0 + kk + 16);
    v16bf af1n = load_frag(A + aoff1 + kk, A + aoff1 + kk + 16);
    v16bf wfn[4];
#pragma unroll
    for (int j = 0; j < 4; ++j) wfn[j] = load_frag(W + woff[j] + kk, W + woff[j] + kk + 16);
#pragma unroll
    for (int j = 0; j < 4; ++j) {
      acc[0][j].v = WMMA_BF16(af0, wf[j], acc[0][j].v);
      acc[1][j].v = WMMA_BF16(af1, wf[j], acc[1][j].v);
    }
    af0 = af0n; af1 = af1n;
#pragma unroll
    for (int j = 0; j < 4; ++j) wf[j] = wfn[j];
  }
#pragma unroll
  for (int j = 0; j < 4; ++j) {
    acc[0][j].v = WMMA_BF16(af0, wf[j], acc[0][j].v);
    acc[1][j].v = WMMA_BF16(af1, wf[j], acc[1][j].v);
  }

#pragma unroll
  for (int i = 0; i < 2; ++i)
#pragma unroll
    for (int j = 0; j < 4; ++j)
#pragma unroll
      for (int r = 0; r < 8; ++r) {
        int row = mlo + i * 16 + r + lh * 8;
        int col = nlo + j * 16 + ll;
        float v = acc[i][j].f[r] + bias[col];
        if (OUT_F32) ((float*)Out)[(size_t)row * N + col] = v;
        else         ((bf16*)Out)[(size_t)row * N + col] = f2b(v);
      }
}

// ---------------------------------------------------------------- RoPE + pack heads
__global__ void pack_rope(const bf16* __restrict__ qkv, bf16* __restrict__ Qo,
                          bf16* __restrict__ Ko, bf16* __restrict__ Vt) {
  const int T = 2048, Hh = 16, Dd = 128, C3 = 6144;
  int tid = blockIdx.x * blockDim.x + threadIdx.x;   // B*H*T*(D/2) = 2^22
  int i = tid & 63;
  int t = (tid >> 6) & 2047;
  int h = (tid >> 17) & 15;
  int b = tid >> 21;
  size_t row = (size_t)(b * T + t) * C3;
  int de = 2 * i;

  float theta = __expf(-(float)de * (9.210340371976184f / 128.f)); // 10000^(-de/128)
  float fr = (float)t * theta;
  float sn, cs;
  __sincosf(fr, &sn, &cs);

  const float scale = 0.08838834764831845f;  // 1/sqrt(128)
  size_t bh = (size_t)(b * Hh + h);
  size_t qidx = (bh * T + t) * Dd + de;

  float qe = b2f(qkv[row + h * 128 + de]);
  float qo = b2f(qkv[row + h * 128 + de + 1]);
  Qo[qidx]     = f2b((qe * cs - qo * sn) * scale);
  Qo[qidx + 1] = f2b((qe * sn + qo * cs) * scale);

  float ke = b2f(qkv[row + 2048 + h * 128 + de]);
  float ko = b2f(qkv[row + 2048 + h * 128 + de + 1]);
  Ko[qidx]     = f2b(ke * cs - ko * sn);
  Ko[qidx + 1] = f2b(ke * sn + ko * cs);

  bf16 ve = qkv[row + 4096 + h * 128 + de];
  bf16 vo = qkv[row + 4096 + h * 128 + de + 1];
  Vt[(bh * Dd + de) * T + t]     = ve;
  Vt[(bh * Dd + de + 1) * T + t] = vo;
}

// ---------------------------------------------------------------- Flash attention
// Block = 4 waves / 4 q-tiles. K/V tiles (shared by all 4 waves) staged into LDS
// by the Tensor Data Mover, double-buffered, overlapped with the current tile's WMMAs.
__global__ __launch_bounds__(128, 2) void flash_attn(
    const bf16* __restrict__ Q, const bf16* __restrict__ Kk,
    const bf16* __restrict__ Vt, bf16* __restrict__ Y) {
  const int T = 2048, Dd = 128, Cc = 2048;
  __shared__ __align__(16) bf16 kbuf[2][32 * 128];           // [key][d]
  __shared__ __align__(16) bf16 vbuf[2][128 * 32];           // [d][key]
  __shared__ __align__(16) unsigned short lds_p[4][16 * 32];

  const int lane = threadIdx.x & 31, wave = threadIdx.x >> 5;
  const int lh = lane >> 4, ll = lane & 15;
  const int bh = blockIdx.y;
  const int b = bh >> 4, h = bh & 15;
  const int qlo = (blockIdx.x * 4 + wave) * 16;

  const bf16* Qb = Q  + (size_t)bh * T * Dd;
  const bf16* Kb = Kk + (size_t)bh * T * Dd;
  const bf16* Vb = Vt + (size_t)bh * Dd * T;

  v16bf qf[4];
  {
    const int q0 = (qlo + ll) * Dd + lh * 8;
#pragma unroll
    for (int c = 0; c < 4; ++c)
      qf[c] = load_frag(Qb + q0 + c * 32, Qb + q0 + c * 32 + 16);
  }

  Acc o[8];
#pragma unroll
  for (int dc = 0; dc < 8; ++dc)
#pragma unroll
    for (int r = 0; r < 8; ++r) o[dc].f[r] = 0.f;
  float mrow[8], lrow[8];
#pragma unroll
  for (int r = 0; r < 8; ++r) { mrow[r] = -1e30f; lrow[r] = 0.f; }

  unsigned short* lp = lds_p[wave];
  const int kmax = blockIdx.x * 64 + 63;     // block-uniform
  const int niter = kmax / 32 + 1;

  if (wave == 0) {                            // stage tile 0 into buffer 0
    tdm_load_2d(lds_off(&kbuf[0][0]), Kb, 128u, 32u, 128u);
    tdm_load_2d(lds_off(&vbuf[0][0]), Vb, 32u, 128u, (unsigned)T);
  }

  for (int it = 0; it < niter; ++it) {
    const int klo = it * 32;
    const int cur = it & 1;
    if (wave == 0) {
      if (it + 1 < niter) {                   // prefetch next tile into other buffer
        const bf16* kn = Kb + (klo + 32) * Dd;
        const bf16* vn = Vb + (klo + 32);
        tdm_load_2d(lds_off(&kbuf[cur ^ 1][0]), kn, 128u, 32u, 128u);
        tdm_load_2d(lds_off(&vbuf[cur ^ 1][0]), vn, 32u, 128u, (unsigned)T);
        __builtin_amdgcn_s_wait_tensorcnt(2); // current tile's 2 TDM ops done
      } else {
        __builtin_amdgcn_s_wait_tensorcnt(0);
      }
    }
    __syncthreads();                          // publish current tile to all waves

    const bf16* kb = kbuf[cur];
    const bf16* vb = vbuf[cur];

    Acc s0, s1;
#pragma unroll
    for (int r = 0; r < 8; ++r) { s0.f[r] = 0.f; s1.f[r] = 0.f; }
#pragma unroll
    for (int c = 0; c < 4; ++c) {
      v16bf b0 = load_frag(kb + ll * 128 + c * 32 + lh * 8,
                           kb + ll * 128 + c * 32 + lh * 8 + 16);
      v16bf b1 = load_frag(kb + (ll + 16) * 128 + c * 32 + lh * 8,
                           kb + (ll + 16) * 128 + c * 32 + lh * 8 + 16);
      s0.v = WMMA_BF16(qf[c], b0, s0.v);
      s1.v = WMMA_BF16(qf[c], b1, s1.v);
    }

    if (klo + 31 > qlo) {   // diagonal / fully-masked tiles
#pragma unroll
      for (int r = 0; r < 8; ++r) {
        int rr = qlo + r + lh * 8;
        int c0 = klo + ll;
        if (c0 > rr)      s0.f[r] = -1e30f;
        if (c0 + 16 > rr) s1.f[r] = -1e30f;
      }
    }

    float corr[8];
#pragma unroll
    for (int r = 0; r < 8; ++r) {
      float mx = half_max(fmaxf(s0.f[r], s1.f[r]));
      float mn = fmaxf(mrow[r], mx);
      float p0 = __expf(s0.f[r] - mn);
      float p1 = __expf(s1.f[r] - mn);
      float sm = half_sum(p0 + p1);
      corr[r] = __expf(mrow[r] - mn);
      lrow[r] = lrow[r] * corr[r] + sm;
      mrow[r] = mn;
      int rowl = r + lh * 8;
      lp[rowl * 32 + ll]      = __builtin_bit_cast(unsigned short, f2b(p0));
      lp[rowl * 32 + 16 + ll] = __builtin_bit_cast(unsigned short, f2b(p1));
    }

    __builtin_amdgcn_wave_barrier();
    asm volatile("s_wait_dscnt 0" ::: "memory");   // in-wave cross-lane LDS visibility

    // P in A-fragment layout (row = ll, K chunks by lane half)
    v16bf pf = load_frag((const bf16*)&lp[ll * 32 + lh * 8],
                         (const bf16*)&lp[ll * 32 + 16 + lh * 8]);
#pragma unroll
    for (int dc = 0; dc < 8; ++dc) {
#pragma unroll
      for (int r = 0; r < 8; ++r) o[dc].f[r] *= corr[r];
      v16bf bv = load_frag(vb + (dc * 16 + ll) * 32 + lh * 8,
                           vb + (dc * 16 + ll) * 32 + lh * 8 + 16);
      o[dc].v = WMMA_BF16(pf, bv, o[dc].v);
    }

    __syncthreads();                          // all reads done before next TDM overwrite
  }

  float rl[8];
#pragma unroll
  for (int r = 0; r < 8; ++r) rl[r] = 1.f / lrow[r];
#pragma unroll
  for (int dc = 0; dc < 8; ++dc)
#pragma unroll
    for (int r = 0; r < 8; ++r) {
      int t = qlo + r + lh * 8;
      int col = h * Dd + dc * 16 + ll;
      Y[((size_t)b * T + t) * Cc + col] = f2b(o[dc].f[r] * rl[r]);
    }
}

// ---------------------------------------------------------------- launch
extern "C" void kernel_launch(void* const* d_in, const int* in_sizes, int n_in,
                              void* d_out, int out_size, void* d_ws, size_t ws_size,
                              hipStream_t stream) {
  (void)in_sizes; (void)n_in; (void)out_size; (void)ws_size;
  const int Bb = 2, Tt = 2048, Cc = 2048, Hh = 16, Dd = 128;
  const int M = Bb * Tt;      // 4096
  const int N3 = 3 * Cc;      // 6144

  const float* x      = (const float*)d_in[0];
  const float* w_att  = (const float*)d_in[1];
  const float* b_att  = (const float*)d_in[2];
  const float* w_proj = (const float*)d_in[3];
  const float* b_proj = (const float*)d_in[4];

  char* ws = (char*)d_ws;
  size_t off = 0;
  auto alloc = [&](size_t bytes) {
    void* p = ws + off;
    off += (bytes + 255) & ~(size_t)255;
    return p;
  };
  bf16* x_bf   = (bf16*)alloc((size_t)M * Cc * 2);
  bf16* wa_bf  = (bf16*)alloc((size_t)N3 * Cc * 2);
  bf16* wp_bf  = (bf16*)alloc((size_t)Cc * Cc * 2);
  bf16* qkv_bf = (bf16*)alloc((size_t)M * N3 * 2);
  bf16* q_bf   = (bf16*)alloc((size_t)M * Cc * 2);
  bf16* k_bf   = (bf16*)alloc((size_t)M * Cc * 2);
  bf16* vt_bf  = (bf16*)alloc((size_t)M * Cc * 2);
  bf16* y_bf   = (bf16*)alloc((size_t)M * Cc * 2);

  cast_f32_bf16<<<4096, 256, 0, stream>>>(x, x_bf, M * Cc);
  cast_f32_bf16<<<4096, 256, 0, stream>>>(w_att, wa_bf, N3 * Cc);
  cast_f32_bf16<<<4096, 256, 0, stream>>>(w_proj, wp_bf, Cc * Cc);

  gemm_bf16_wmma<false><<<dim3(N3 / 64, M / 128), 128, 0, stream>>>(
      x_bf, wa_bf, b_att, qkv_bf, M, N3, Cc);

  pack_rope<<<(Bb * Hh * Tt * (Dd / 2)) / 256, 256, 0, stream>>>(qkv_bf, q_bf, k_bf, vt_bf);

  flash_attn<<<dim3(Tt / 64, Bb * Hh), 128, 0, stream>>>(q_bf, k_bf, vt_bf, y_bf);

  gemm_bf16_wmma<true><<<dim3(Cc / 64, M / 128), 128, 0, stream>>>(
      y_bf, wp_bf, b_proj, d_out, M, Cc, Cc);
}
